// SelfAttention2d_996432413325
// MI455X (gfx1250) — matrix operand order repaired and checked by reading, not compiled, not run on hardware
//
#include <hip/hip_runtime.h>

#define B_     4
#define C_     256
#define N_     4096
#define HEADS_ 4
#define HD_    64
#define G_     8

typedef __attribute__((ext_vector_type(16))) __bf16 v16bf;
typedef __attribute__((ext_vector_type(8)))  float  v8f;

__device__ __forceinline__ v8f wmma_bf16f32(v16bf a, v16bf b, v8f c) {
  // D = A(16x32 bf16) x B(32x16 bf16) + C(16x16 f32)
  return __builtin_amdgcn_wmma_f32_16x16x32_bf16(false, a, false, b, (short)0, c,
                                                 false, false);
}

// Async global->LDS 128-bit copy (gfx1250, tracked by ASYNCcnt).
__device__ __forceinline__ void async_copy_b128(unsigned lds_off, const void* gptr) {
  asm volatile("global_load_async_to_lds_b128 %0, %1, off"
               :: "v"(lds_off), "v"((unsigned long long)(size_t)gptr)
               : "memory");
}

// ---------------------------------------------------------------------------
// GroupNorm statistics: one block per (b, group); 32 channels * 4096 px each.
// ---------------------------------------------------------------------------
__global__ __launch_bounds__(256) void gn_stats_kernel(const float* __restrict__ x,
                                                       float* __restrict__ stats) {
  const int bg = blockIdx.x;                       // b * G_ + g
  const float* xb = x + (size_t)bg * (32 * N_);    // channels of a group contiguous
  float s = 0.f, ss = 0.f;
  for (int i = threadIdx.x; i < 32 * N_; i += 256) {
    float t = xb[i];
    s += t; ss += t * t;
  }
  __shared__ float rs[256], rq[256];
  rs[threadIdx.x] = s; rq[threadIdx.x] = ss;
  __syncthreads();
  for (int off = 128; off > 0; off >>= 1) {
    if (threadIdx.x < off) {
      rs[threadIdx.x] += rs[threadIdx.x + off];
      rq[threadIdx.x] += rq[threadIdx.x + off];
    }
    __syncthreads();
  }
  if (threadIdx.x == 0) {
    const float inv_n = 1.0f / (32.0f * (float)N_);
    float mean = rs[0] * inv_n;
    float var  = rq[0] * inv_n - mean * mean;
    stats[bg * 2 + 0] = mean;
    stats[bg * 2 + 1] = rsqrtf(var + 1e-5f);
  }
}

// ---------------------------------------------------------------------------
// GroupNorm apply -> bf16 h (B, C, N)
// ---------------------------------------------------------------------------
__global__ __launch_bounds__(256) void gn_apply_kernel(const float* __restrict__ x,
                                                       const float* __restrict__ w,
                                                       const float* __restrict__ bias,
                                                       const float* __restrict__ stats,
                                                       __bf16* __restrict__ h) {
  const int i = blockIdx.x * 256 + threadIdx.x;    // < B*C*N = 4,194,304
  const int c  = (i >> 12) & 255;                  // N_=4096=2^12
  const int bg = i >> 17;                          // / (32*4096)
  const float mean = stats[bg * 2 + 0];
  const float rstd = stats[bg * 2 + 1];
  h[i] = (__bf16)((x[i] - mean) * rstd * w[c] + bias[c]);
}

// ---------------------------------------------------------------------------
// Weight conversion to bf16
// ---------------------------------------------------------------------------
__global__ __launch_bounds__(256) void cvt_w_kernel(const float* __restrict__ qkvw,
                                                    const float* __restrict__ projw,
                                                    __bf16* __restrict__ wq,
                                                    __bf16* __restrict__ wp) {
  const int i = blockIdx.x * 256 + threadIdx.x;
  if (i < 768 * 256) wq[i] = (__bf16)qkvw[i];
  if (i < 256 * 256) wp[i] = (__bf16)projw[i];
}

// ---------------------------------------------------------------------------
// QKV GEMM: qkv[o,n] = sum_c W[o,c] * h[c,n] + b[o]; split into q/k/v (B,C,N) bf16
// One wave per 16x16 output tile.
// ---------------------------------------------------------------------------
__global__ __launch_bounds__(32) void qkv_gemm_kernel(const __bf16* __restrict__ w,
                                                      const float* __restrict__ bias,
                                                      const __bf16* __restrict__ h,
                                                      __bf16* __restrict__ q,
                                                      __bf16* __restrict__ k,
                                                      __bf16* __restrict__ v) {
  const int n0 = blockIdx.x * 16;
  const int o0 = blockIdx.y * 16;
  const int b  = blockIdx.z;
  const int lane  = threadIdx.x;
  const int row   = lane & 15;
  const int khalf = lane >> 4;
  const int koff  = khalf * 16;
  const __bf16* hb = h + (size_t)b * C_ * N_;

  v8f acc = {};
  for (int cb = 0; cb < C_; cb += 32) {
    v16bf a, bm;
    const __bf16* wr = w + (o0 + row) * C_ + cb + khalf * 8;
#pragma unroll
    for (int j = 0; j < 8; ++j) { a[j] = wr[j]; a[j + 8] = wr[16 + j]; }
#pragma unroll
    for (int e = 0; e < 16; ++e) bm[e] = hb[(cb + khalf * 8 + (e & 7) + ((e >> 3) << 4)) * N_ + n0 + row];
    acc = wmma_bf16f32(a, bm, acc);
  }
  const int part = o0 >> 8;                        // 0:q 1:k 2:v (tile never crosses)
  __bf16* dst = (part == 0) ? q : ((part == 1) ? k : v);
#pragma unroll
  for (int r = 0; r < 8; ++r) {
    const int o  = o0 + r + 8 * khalf;
    const int cl = o & 255;
    dst[((size_t)b * C_ + cl) * N_ + n0 + row] = (__bf16)(acc[r] + bias[o]);
  }
}

// ---------------------------------------------------------------------------
// Flash attention: 4 waves/block share one (b, head); each wave owns a 16-row
// Q tile. KV is streamed in 32-column blocks staged into LDS with async
// global->LDS b128 copies shared by all 4 waves.
// ---------------------------------------------------------------------------
__global__ __launch_bounds__(128) void attn_kernel(const __bf16* __restrict__ q,
                                                   const __bf16* __restrict__ k,
                                                   const __bf16* __restrict__ v,
                                                   __bf16* __restrict__ ao) {
  __shared__ __bf16 kbuf[64 * 32];                 // K block, [c][m], 4 KB
  __shared__ __bf16 vbuf[64 * 32];                 // V block, [c][m], 4 KB
  __shared__ __bf16 pbuf[4 * 16 * 32];             // per-wave 16x32 P tile
  const int tid   = threadIdx.x;
  const int lane  = tid & 31;
  const int wid   = tid >> 5;
  const int row   = lane & 15;
  const int khalf = lane >> 4;
  const int koff  = khalf * 16;

  const int bidx = blockIdx.x;                     // B*HEADS*64 = 1024 blocks
  const int b  = bidx >> 8;
  const int hh = (bidx >> 6) & 3;
  const int tb = bidx & 63;
  const int n0 = (tb * 4 + wid) * 16;

  const int base = (b * C_ + hh * HD_) * N_;
  const __bf16* qh = q + base;
  const __bf16* kh = k + base;
  const __bf16* vh = v + base;

  // A-fragments of Q^T (n x c), pre-scaled by 1/sqrt(hd)=0.125
  v16bf qa0, qa1;
#pragma unroll
  for (int j = 0; j < 8; ++j) {
    const int c0 = khalf * 16 + j;
    const int c1 = khalf * 16 + 8 + j;
    qa0[j]     = (__bf16)((float)qh[c0 * N_ + n0 + row] * 0.125f);
    qa0[j + 8] = (__bf16)((float)qh[c1 * N_ + n0 + row] * 0.125f);
    qa1[j]     = (__bf16)((float)qh[(c0 + 32) * N_ + n0 + row] * 0.125f);
    qa1[j + 8] = (__bf16)((float)qh[(c1 + 32) * N_ + n0 + row] * 0.125f);
  }

  v8f o0v = {}, o1v = {}, o2v = {}, o3v = {};
  float mrow[8], lrow[8];
#pragma unroll
  for (int r = 0; r < 8; ++r) { mrow[r] = -3.0e38f; lrow[r] = 0.0f; }

  __bf16* pb = pbuf + wid * (16 * 32);
  const unsigned kb0 = (unsigned)(size_t)kbuf;     // LDS byte offsets
  const unsigned vb0 = (unsigned)(size_t)vbuf;

  for (int m0 = 0; m0 < N_; m0 += 32) {
    // ---- Cooperative async staging of K and V blocks (512 x b128) ----------
#pragma unroll
    for (int i = 0; i < 4; ++i) {
      const int r   = tid + (i & 1) * 128;         // 0..255 within K or V half
      const int c   = r >> 2;                      // channel row 0..63
      const int seg = (r & 3) * 8;                 // 8 bf16 (16B) segment
      const unsigned loff = (unsigned)(c * 32 + seg) * 2u;
      if (i < 2) async_copy_b128(kb0 + loff, kh + c * N_ + m0 + seg);
      else       async_copy_b128(vb0 + loff, vh + c * N_ + m0 + seg);
    }
    asm volatile("s_wait_asynccnt 0" ::: "memory");
    __syncthreads();

    // ---- B-fragments of K (c x m) from LDS: two m-tiles x two c-halves -----
    v16bf kf00, kf01, kf10, kf11;
#pragma unroll
    for (int e = 0; e < 16; ++e) {
      kf00[e] = kbuf[(koff + e) * 32 + row];
      kf01[e] = kbuf[(32 + koff + e) * 32 + row];
      kf10[e] = kbuf[(koff + e) * 32 + 16 + row];
      kf11[e] = kbuf[(32 + koff + e) * 32 + 16 + row];
    }
    v8f s0 = {}, s1 = {};
    s0 = wmma_bf16f32(qa0, kf00, s0);
    s0 = wmma_bf16f32(qa1, kf01, s0);
    s1 = wmma_bf16f32(qa0, kf10, s1);
    s1 = wmma_bf16f32(qa1, kf11, s1);

    // ---- Online softmax over the 32 new columns ---------------------------
    float p0[8], p1[8], corr[8];
#pragma unroll
    for (int r = 0; r < 8; ++r) {
      float mx = fmaxf(s0[r], s1[r]);
      mx = fmaxf(mx, __shfl_xor(mx, 1));
      mx = fmaxf(mx, __shfl_xor(mx, 2));
      mx = fmaxf(mx, __shfl_xor(mx, 4));
      mx = fmaxf(mx, __shfl_xor(mx, 8));
      const float mnew = fmaxf(mrow[r], mx);
      corr[r] = __expf(mrow[r] - mnew);
      const float e0 = __expf(s0[r] - mnew);
      const float e1 = __expf(s1[r] - mnew);
      float ls = e0 + e1;
      ls += __shfl_xor(ls, 1);
      ls += __shfl_xor(ls, 2);
      ls += __shfl_xor(ls, 4);
      ls += __shfl_xor(ls, 8);
      lrow[r] = lrow[r] * corr[r] + ls;
      mrow[r] = mnew;
      p0[r] = e0; p1[r] = e1;
    }
#pragma unroll
    for (int r = 0; r < 8; ++r) {
      o0v[r] *= corr[r]; o1v[r] *= corr[r]; o2v[r] *= corr[r]; o3v[r] *= corr[r];
    }

    // ---- Stash P (16x32) row-major in per-wave LDS, reload as A-fragment --
#pragma unroll
    for (int r = 0; r < 8; ++r) {
      const int prow = r + 8 * khalf;
      pb[prow * 32 + row]      = (__bf16)p0[r];
      pb[prow * 32 + 16 + row] = (__bf16)p1[r];
    }
    asm volatile("s_wait_dscnt 0" ::: "memory");
    v16bf pa;
#pragma unroll
    for (int j = 0; j < 8; ++j) {
      pa[j]     = pb[row * 32 + koff + j];
      pa[j + 8] = pb[row * 32 + koff + 8 + j];
    }

    // ---- O^T(n x c) += P(n x m) @ V^T(m x c); V fragments contiguous in LDS
    v16bf vf0, vf1, vf2, vf3;
#pragma unroll
    for (int e = 0; e < 16; ++e) {
      vf0[e] = vbuf[(0 * 16 + row) * 32 + koff + e];
      vf1[e] = vbuf[(1 * 16 + row) * 32 + koff + e];
      vf2[e] = vbuf[(2 * 16 + row) * 32 + koff + e];
      vf3[e] = vbuf[(3 * 16 + row) * 32 + koff + e];
    }
    o0v = wmma_bf16f32(pa, vf0, o0v);
    o1v = wmma_bf16f32(pa, vf1, o1v);
    o2v = wmma_bf16f32(pa, vf2, o2v);
    o3v = wmma_bf16f32(pa, vf3, o3v);

    __syncthreads();                               // LDS reuse guard
  }

  // Normalize rows and store O^T into ao (c-major, bf16).
#pragma unroll
  for (int r = 0; r < 8; ++r) {
    const float inv = 1.0f / lrow[r];
    const int n = n0 + r + 8 * khalf;
    ao[base + (0 * 16 + row) * N_ + n] = (__bf16)(o0v[r] * inv);
    ao[base + (1 * 16 + row) * N_ + n] = (__bf16)(o1v[r] * inv);
    ao[base + (2 * 16 + row) * N_ + n] = (__bf16)(o2v[r] * inv);
    ao[base + (3 * 16 + row) * N_ + n] = (__bf16)(o3v[r] * inv);
  }
}

// ---------------------------------------------------------------------------
// Proj GEMM + bias + residual -> fp32 output
// ---------------------------------------------------------------------------
__global__ __launch_bounds__(32) void proj_gemm_kernel(const __bf16* __restrict__ w,
                                                       const float* __restrict__ bias,
                                                       const __bf16* __restrict__ ao,
                                                       const float* __restrict__ x,
                                                       float* __restrict__ out) {
  const int n0 = blockIdx.x * 16;
  const int o0 = blockIdx.y * 16;
  const int b  = blockIdx.z;
  const int lane  = threadIdx.x;
  const int row   = lane & 15;
  const int khalf = lane >> 4;
  const int koff  = khalf * 16;
  const __bf16* ab = ao + (size_t)b * C_ * N_;

  v8f acc = {};
  for (int cb = 0; cb < C_; cb += 32) {
    v16bf a, bm;
    const __bf16* wr = w + (o0 + row) * C_ + cb + khalf * 8;
#pragma unroll
    for (int j = 0; j < 8; ++j) { a[j] = wr[j]; a[j + 8] = wr[16 + j]; }
#pragma unroll
    for (int e = 0; e < 16; ++e) bm[e] = ab[(cb + khalf * 8 + (e & 7) + ((e >> 3) << 4)) * N_ + n0 + row];
    acc = wmma_bf16f32(a, bm, acc);
  }
#pragma unroll
  for (int r = 0; r < 8; ++r) {
    const int o   = o0 + r + 8 * khalf;
    const int idx = (b * C_ + o) * N_ + n0 + row;
    out[idx] = acc[r] + bias[o] + x[idx];
  }
}

// ---------------------------------------------------------------------------
extern "C" void kernel_launch(void* const* d_in, const int* in_sizes, int n_in,
                              void* d_out, int out_size, void* d_ws, size_t ws_size,
                              hipStream_t stream) {
  (void)in_sizes; (void)n_in; (void)out_size; (void)ws_size;
  const float* x      = (const float*)d_in[0];
  const float* norm_w = (const float*)d_in[1];
  const float* norm_b = (const float*)d_in[2];
  const float* qkv_w  = (const float*)d_in[3];
  const float* qkv_b  = (const float*)d_in[4];
  const float* proj_w = (const float*)d_in[5];
  const float* proj_b = (const float*)d_in[6];
  float* out = (float*)d_out;

  const size_t plane = (size_t)B_ * C_ * N_;       // 4,194,304 elements
  char* p = (char*)d_ws;
  float*  stats = (float*)p;  p += 256;
  __bf16* h     = (__bf16*)p; p += plane * 2;
  __bf16* qb    = (__bf16*)p; p += plane * 2;
  __bf16* kb    = (__bf16*)p; p += plane * 2;
  __bf16* vb    = (__bf16*)p; p += plane * 2;
  __bf16* aob   = (__bf16*)p; p += plane * 2;
  __bf16* wq    = (__bf16*)p; p += (size_t)768 * 256 * 2;
  __bf16* wp    = (__bf16*)p;

  cvt_w_kernel<<<768, 256, 0, stream>>>(qkv_w, proj_w, wq, wp);
  gn_stats_kernel<<<B_ * G_, 256, 0, stream>>>(x, stats);
  gn_apply_kernel<<<(int)(plane / 256), 256, 0, stream>>>(x, norm_w, norm_b, stats, h);
  qkv_gemm_kernel<<<dim3(N_ / 16, 48, B_), 32, 0, stream>>>(wq, qkv_b, h, qb, kb, vb);
  attn_kernel<<<B_ * HEADS_ * (N_ / 16) / 4, 128, 0, stream>>>(qb, kb, vb, aob);
  proj_gemm_kernel<<<dim3(N_ / 16, 16, B_), 32, 0, stream>>>(wp, proj_b, aob, x, out);
}
